// NeuralODEModel_83975200571469
// MI455X (gfx1250) — matrix operand order repaired and checked by reading, not compiled, not run on hardware
//
#include <hip/hip_runtime.h>

typedef _Float16 half_t;
typedef __attribute__((ext_vector_type(16))) _Float16 v16h;
typedef __attribute__((ext_vector_type(8)))  _Float16 v8h;
typedef __attribute__((ext_vector_type(4)))  _Float16 v4h;
typedef __attribute__((ext_vector_type(8)))  float    v8f;
typedef __attribute__((ext_vector_type(4)))  float    v4f;

#define DEV_INLINE __device__ __forceinline__

// ---------------- Dormand-Prince 5(4) coefficients ----------------
static __device__ constexpr float AC[6][5] = {
  {0.f, 0.f, 0.f, 0.f, 0.f},
  {1.f/5.f, 0.f, 0.f, 0.f, 0.f},
  {3.f/40.f, 9.f/40.f, 0.f, 0.f, 0.f},
  {44.f/45.f, -56.f/15.f, 32.f/9.f, 0.f, 0.f},
  {19372.f/6561.f, -25360.f/2187.f, 64448.f/6561.f, -212.f/729.f, 0.f},
  {9017.f/3168.f, -355.f/33.f, 46732.f/5247.f, 49.f/176.f, -5103.f/18656.f}
};
static __device__ constexpr float BC1 = 35.f/384.f, BC3 = 500.f/1113.f,
                                  BC4 = 125.f/192.f, BC5 = -2187.f/6784.f,
                                  BC6 = 11.f/84.f;

// ---------------- WMMA helpers ----------------
DEV_INLINE v16h cat8(v8h lo, v8h hi) {
  return __builtin_shufflevector(lo, hi, 0,1,2,3,4,5,6,7,8,9,10,11,12,13,14,15);
}

DEV_INLINE v8f wmma_f16(v16h a, v16h b, v8f c) {
  return __builtin_amdgcn_wmma_f32_16x16x32_f16(false, a, false, b, (short)0, c,
                                                false, false);
}

// A-fragment: lane L holds row M=L%16; K-subrange selected by L/16.
template<int KC>
DEV_INLINE void load_afrags(const half_t* A, int rowStride, v16h* afr,
                            int laneRow, int laneHi) {
#pragma unroll
  for (int kc = 0; kc < KC; ++kc) {
    const half_t* p = A + laneRow * rowStride + kc * 32 + laneHi * 16;
    v8h lo = *(const v8h*)p;
    v8h hi = *(const v8h*)(p + 8);
    afr[kc] = cat8(lo, hi);
  }
}

// B-fragment from row-major [N][K] f16 weights: lane L -> N = nt*16+L%16,
// K = kc*32 + (L/16)*16 + i  (16 consecutive halves)
template<int KIN>
DEV_INLINE v16h load_bfrag(const half_t* W, int nt, int kc, int laneRow, int laneHi) {
  const half_t* p = W + (nt * 16 + laneRow) * KIN + kc * 32 + laneHi * 16;
  v8h lo = *(const v8h*)p;
  v8h hi = *(const v8h*)(p + 8);
  return cat8(lo, hi);
}

// One MLP layer (staging -> staging), relu; weights in LDS f16, bias in regs.
// Output tiles processed in pairs: two independent WMMA chains fill each
// other's WMMA->VALU hazard slots (avoids 4x v_nop per tile).
template<int KIN, int NOUT>
DEV_INLINE void layer_relu(half_t* st, const half_t* W, const float* biasReg,
                           int laneRow, int laneHi) {
  constexpr int KC = KIN / 32;
  v16h afr[KC];
  load_afrags<KC>(st, 128, afr, laneRow, laneHi);   // read all A before overwrite
#pragma unroll
  for (int np = 0; np < NOUT / 32; ++np) {
    const int nt0 = 2 * np, nt1 = 2 * np + 1;
    v8f acc0 = {}, acc1 = {};
#pragma unroll
    for (int kc = 0; kc < KC; ++kc) {
      acc0 = wmma_f16(afr[kc], load_bfrag<KIN>(W, nt0, kc, laneRow, laneHi), acc0);
      acc1 = wmma_f16(afr[kc], load_bfrag<KIN>(W, nt1, kc, laneRow, laneHi), acc1);
    }
#pragma unroll
    for (int v = 0; v < 8; ++v) {
      float r0 = fmaxf(acc0[v] + biasReg[nt0], 0.f);
      float r1 = fmaxf(acc1[v] + biasReg[nt1], 0.f);
      // D layout: M = v + 8*laneHi, N = nt*16 + laneRow
      st[(8 * laneHi + v) * 128 + nt0 * 16 + laneRow] = (half_t)r0;
      st[(8 * laneHi + v) * 128 + nt1 * 16 + laneRow] = (half_t)r1;
    }
  }
}

// ---------------- ODE kernel ----------------
struct OdeCtx {
  half_t *W1, *W2, *W3, *W4, *st;
  float *zw, *kw;                      // kw: [6][1024] per wave
  float b1r[8], b2r[8], b3r[8], b4r[4];
  int lane, laneRow, laneHi;
};

template<int S>
DEV_INLINE void do_stage(const OdeCtx& c, float dt) {
  // build stage input (16 rows x 64 cols, row stride 128 halves), vectorized x4
#pragma unroll
  for (int t = 0; t < 8; ++t) {
    int off = c.lane * 32 + t * 4;
    v4f v = *(const v4f*)&c.zw[off];
#pragma unroll
    for (int j = 0; j < S; ++j)
      v += (dt * AC[S][j]) * *(const v4f*)&c.kw[j * 1024 + off];
    *(v4h*)&c.st[(off >> 6) * 128 + (off & 63)] = __builtin_convertvector(v, v4h);
  }
  layer_relu<64, 128>(c.st, c.W1, c.b1r, c.laneRow, c.laneHi);
  layer_relu<128, 128>(c.st, c.W2, c.b2r, c.laneRow, c.laneHi);
  layer_relu<128, 128>(c.st, c.W3, c.b3r, c.laneRow, c.laneHi);
  // layer 4 (128 -> 64), no relu; result -> k[S]; tiles in pairs
  v16h afr[4];
  load_afrags<4>(c.st, 128, afr, c.laneRow, c.laneHi);
#pragma unroll
  for (int np = 0; np < 2; ++np) {
    const int nt0 = 2 * np, nt1 = 2 * np + 1;
    v8f acc0 = {}, acc1 = {};
#pragma unroll
    for (int kc = 0; kc < 4; ++kc) {
      acc0 = wmma_f16(afr[kc], load_bfrag<128>(c.W4, nt0, kc, c.laneRow, c.laneHi), acc0);
      acc1 = wmma_f16(afr[kc], load_bfrag<128>(c.W4, nt1, kc, c.laneRow, c.laneHi), acc1);
    }
#pragma unroll
    for (int v = 0; v < 8; ++v) {
      int base = S * 1024 + (8 * c.laneHi + v) * 64 + c.laneRow;
      c.kw[base + nt0 * 16] = acc0[v] + c.b4r[nt0];
      c.kw[base + nt1 * 16] = acc1[v] + c.b4r[nt1];
    }
  }
}

__global__ void k_ode(const float* __restrict__ z0, float* __restrict__ zT,
                      const float* __restrict__ ow1, const float* __restrict__ ob1,
                      const float* __restrict__ ow2, const float* __restrict__ ob2,
                      const float* __restrict__ ow3, const float* __restrict__ ob3,
                      const float* __restrict__ ow4, const float* __restrict__ ob4) {
  extern __shared__ char smem[];
  half_t* W1 = (half_t*)smem;            // 128x64
  half_t* W2 = W1 + 8192;                // 128x128
  half_t* W3 = W2 + 16384;               // 128x128
  half_t* W4 = W3 + 16384;               // 64x128
  float*  zb = (float*)(W4 + 8192);      // 4 waves * 16x64
  float*  kb = zb + 4 * 1024;            // 4 waves * 6 * 16x64
  half_t* st = (half_t*)(kb + 4 * 6 * 1024);  // 4 waves * 16x128

  const int tid = threadIdx.x;
  const int wave = tid >> 5, lane = tid & 31;
  const int laneRow = lane & 15, laneHi = lane >> 4;

  for (int i = tid * 4; i < 128 * 64;  i += 512)
    *(v4h*)&W1[i] = __builtin_convertvector(*(const v4f*)&ow1[i], v4h);
  for (int i = tid * 4; i < 128 * 128; i += 512)
    *(v4h*)&W2[i] = __builtin_convertvector(*(const v4f*)&ow2[i], v4h);
  for (int i = tid * 4; i < 128 * 128; i += 512)
    *(v4h*)&W3[i] = __builtin_convertvector(*(const v4f*)&ow3[i], v4h);
  for (int i = tid * 4; i < 64 * 128;  i += 512)
    *(v4h*)&W4[i] = __builtin_convertvector(*(const v4f*)&ow4[i], v4h);
  __syncthreads();

  OdeCtx c;
  c.W1 = W1; c.W2 = W2; c.W3 = W3; c.W4 = W4;
  c.st = st + wave * 2048;
  c.zw = zb + wave * 1024;
  c.kw = kb + wave * 6 * 1024;
  c.lane = lane; c.laneRow = laneRow; c.laneHi = laneHi;
#pragma unroll
  for (int nt = 0; nt < 8; ++nt) {
    c.b1r[nt] = ob1[nt * 16 + laneRow];
    c.b2r[nt] = ob2[nt * 16 + laneRow];
    c.b3r[nt] = ob3[nt * 16 + laneRow];
  }
#pragma unroll
  for (int nt = 0; nt < 4; ++nt) c.b4r[nt] = ob4[nt * 16 + laneRow];

  const int rowBase = blockIdx.x * 64 + wave * 16;
  const float* zin = z0 + (size_t)rowBase * 64;      // wave chunk is contiguous
#pragma unroll
  for (int t = 0; t < 8; ++t) {
    int off = lane * 32 + t * 4;
    *(v4f*)&c.zw[off] = *(const v4f*)&zin[off];
  }

  const float dt = 1.0f / 32.0f;
  for (int step = 0; step < 32; ++step) {
    do_stage<0>(c, dt);
    do_stage<1>(c, dt);
    do_stage<2>(c, dt);
    do_stage<3>(c, dt);
    do_stage<4>(c, dt);
    do_stage<5>(c, dt);
    // z += dt * (B1 k1 + B3 k3 + B4 k4 + B5 k5 + B6 k6), vectorized x4
#pragma unroll
    for (int t = 0; t < 8; ++t) {
      int off = lane * 32 + t * 4;
      v4f z = *(const v4f*)&c.zw[off];
      z += dt * (BC1 * *(const v4f*)&c.kw[0 * 1024 + off] +
                 BC3 * *(const v4f*)&c.kw[2 * 1024 + off] +
                 BC4 * *(const v4f*)&c.kw[3 * 1024 + off] +
                 BC5 * *(const v4f*)&c.kw[4 * 1024 + off] +
                 BC6 * *(const v4f*)&c.kw[5 * 1024 + off]);
      *(v4f*)&c.zw[off] = z;
    }
  }
  float* zout = zT + (size_t)rowBase * 64;
#pragma unroll
  for (int t = 0; t < 8; ++t) {
    int off = lane * 32 + t * 4;
    *(v4f*)&zout[off] = *(const v4f*)&c.zw[off];
  }
}

// ---------------- encoder conv kernels (scalar) ----------------
__global__ void k_conv1(const float* __restrict__ x, const float* __restrict__ w,
                        const float* __restrict__ bias, half_t* __restrict__ h1) {
  int idx = blockIdx.x * 256 + threadIdx.x;
  if (idx >= 4096 * 16 * 196) return;
  int j = idx % 14, i = (idx / 14) % 14, c = (idx / 196) % 16, b = idx / 3136;
  float wk[9];
#pragma unroll
  for (int t = 0; t < 9; ++t) wk[t] = w[c * 9 + t];
  float bv = bias[c];
  const float* xi = x + (size_t)b * 784;
  float m = -3.0e38f;
#pragma unroll
  for (int di = 0; di < 2; ++di)
#pragma unroll
    for (int dj = 0; dj < 2; ++dj) {
      int y = 2 * i + di, xx = 2 * j + dj;
      float s = bv;
#pragma unroll
      for (int ky = 0; ky < 3; ++ky)
#pragma unroll
        for (int kx = 0; kx < 3; ++kx) {
          int yy = y + ky - 1, xc = xx + kx - 1;
          if (yy >= 0 && yy < 28 && xc >= 0 && xc < 28)
            s += xi[yy * 28 + xc] * wk[ky * 3 + kx];
        }
      m = fmaxf(m, s);
    }
  h1[idx] = (half_t)fmaxf(m, 0.f);
}

__global__ void k_conv2(const half_t* __restrict__ h1, const float* __restrict__ w,
                        const float* __restrict__ bias, half_t* __restrict__ h2) {
  int idx = blockIdx.x * 256 + threadIdx.x;
  if (idx >= 4096 * 32 * 49) return;
  int j = idx % 7, i = (idx / 7) % 7, c = (idx / 49) % 32, b = idx / 1568;
  const half_t* hb = h1 + (size_t)b * 16 * 196;
  const float* wc = w + c * 144;
  float bv = bias[c];
  float m = -3.0e38f;
#pragma unroll
  for (int di = 0; di < 2; ++di)
#pragma unroll
    for (int dj = 0; dj < 2; ++dj) {
      int y = 2 * i + di, xx = 2 * j + dj;
      float s = bv;
      for (int ci = 0; ci < 16; ++ci)
#pragma unroll
        for (int ky = 0; ky < 3; ++ky)
#pragma unroll
          for (int kx = 0; kx < 3; ++kx) {
            int yy = y + ky - 1, xc = xx + kx - 1;
            if (yy >= 0 && yy < 14 && xc >= 0 && xc < 14)
              s += (float)hb[ci * 196 + yy * 14 + xc] * wc[ci * 9 + ky * 3 + kx];
          }
      m = fmaxf(m, s);
    }
  h2[(size_t)b * 1568 + c * 49 + i * 7 + j] = (half_t)fmaxf(m, 0.f);
}

// ---------------- encoder linear kernels (WMMA) ----------------
__global__ void k_lin1(const half_t* __restrict__ h2, const float* __restrict__ ew1,
                       const float* __restrict__ eb1, half_t* __restrict__ u1) {
  const int tid = threadIdx.x;
  const int wave = tid >> 5, lane = tid & 31;
  const int laneRow = lane & 15, laneHi = lane >> 4;
  const int rowBase = blockIdx.x * 128 + wave * 16;

  float br[8];
#pragma unroll
  for (int nt = 0; nt < 8; ++nt) br[nt] = eb1[nt * 16 + laneRow];

  v8f acc[8];
#pragma unroll
  for (int nt = 0; nt < 8; ++nt) acc[nt] = (v8f){};
  for (int kc = 0; kc < 49; ++kc) {
    const half_t* ap = h2 + (size_t)(rowBase + laneRow) * 1568 + kc * 32 + laneHi * 16;
    v16h a = cat8(*(const v8h*)ap, *(const v8h*)(ap + 8));
#pragma unroll
    for (int nt = 0; nt < 8; ++nt) {
      const float* wp = ew1 + (size_t)(nt * 16 + laneRow) * 1568 + kc * 32 + laneHi * 16;
      v16h b;
#pragma unroll
      for (int t = 0; t < 16; ++t) b[t] = (half_t)wp[t];
      acc[nt] = wmma_f16(a, b, acc[nt]);
    }
  }
#pragma unroll
  for (int nt = 0; nt < 8; ++nt)
#pragma unroll
    for (int v = 0; v < 8; ++v) {
      float r = fmaxf(acc[nt][v] + br[nt], 0.f);
      u1[(size_t)(rowBase + 8 * laneHi + v) * 128 + nt * 16 + laneRow] = (half_t)r;
    }
}

__global__ void k_lin2(const half_t* __restrict__ u1, const float* __restrict__ ew2,
                       const float* __restrict__ eb2, float* __restrict__ z0) {
  const int tid = threadIdx.x;
  const int wave = tid >> 5, lane = tid & 31;
  const int laneRow = lane & 15, laneHi = lane >> 4;
  const int rowBase = blockIdx.x * 128 + wave * 16;

  v16h afr[4];
#pragma unroll
  for (int kc = 0; kc < 4; ++kc) {
    const half_t* ap = u1 + (size_t)(rowBase + laneRow) * 128 + kc * 32 + laneHi * 16;
    afr[kc] = cat8(*(const v8h*)ap, *(const v8h*)(ap + 8));
  }
#pragma unroll
  for (int np = 0; np < 2; ++np) {
    const int nt0 = 2 * np, nt1 = 2 * np + 1;
    float bv0 = eb2[nt0 * 16 + laneRow];
    float bv1 = eb2[nt1 * 16 + laneRow];
    v8f acc0 = {}, acc1 = {};
#pragma unroll
    for (int kc = 0; kc < 4; ++kc) {
      v16h b0, b1;
      const float* wp0 = ew2 + (size_t)(nt0 * 16 + laneRow) * 128 + kc * 32 + laneHi * 16;
      const float* wp1 = ew2 + (size_t)(nt1 * 16 + laneRow) * 128 + kc * 32 + laneHi * 16;
#pragma unroll
      for (int t = 0; t < 16; ++t) { b0[t] = (half_t)wp0[t]; b1[t] = (half_t)wp1[t]; }
      acc0 = wmma_f16(afr[kc], b0, acc0);
      acc1 = wmma_f16(afr[kc], b1, acc1);
    }
#pragma unroll
    for (int v = 0; v < 8; ++v) {
      size_t rbase = (size_t)(rowBase + 8 * laneHi + v) * 64 + laneRow;
      z0[rbase + nt0 * 16] = tanhf(acc0[v] + bv0);
      z0[rbase + nt1 * 16] = tanhf(acc1[v] + bv1);
    }
  }
}

// ---------------- classifier (scalar) ----------------
__global__ void k_cls(const float* __restrict__ zT,
                      const float* __restrict__ lw1, const float* __restrict__ lb1,
                      const float* __restrict__ lw2, const float* __restrict__ lb2,
                      float* __restrict__ out) {
  int b = blockIdx.x * 256 + threadIdx.x;
  if (b >= 4096) return;
  const float* zr = zT + (size_t)b * 64;
  float h[64];
  for (int j = 0; j < 64; ++j) {
    float s = lb1[j];
    for (int k = 0; k < 64; ++k) s += zr[k] * lw1[j * 64 + k];
    h[j] = fmaxf(s, 0.f);
  }
  for (int c = 0; c < 10; ++c) {
    float s = lb2[c];
    for (int j = 0; j < 64; ++j) s += h[j] * lw2[c * 64 + j];
    out[(size_t)b * 10 + c] = s;
  }
}

// ---------------- launch ----------------
extern "C" void kernel_launch(void* const* d_in, const int* in_sizes, int n_in,
                              void* d_out, int out_size, void* d_ws, size_t ws_size,
                              hipStream_t stream) {
  (void)in_sizes; (void)n_in; (void)out_size; (void)ws_size;
  const float* x   = (const float*)d_in[0];
  const float* cw1 = (const float*)d_in[1];
  const float* cb1 = (const float*)d_in[2];
  const float* cw2 = (const float*)d_in[3];
  const float* cb2 = (const float*)d_in[4];
  const float* ew1 = (const float*)d_in[5];
  const float* eb1 = (const float*)d_in[6];
  const float* ew2 = (const float*)d_in[7];
  const float* eb2 = (const float*)d_in[8];
  const float* ow1 = (const float*)d_in[9];
  const float* ob1 = (const float*)d_in[10];
  const float* ow2 = (const float*)d_in[11];
  const float* ob2 = (const float*)d_in[12];
  const float* ow3 = (const float*)d_in[13];
  const float* ob3 = (const float*)d_in[14];
  const float* ow4 = (const float*)d_in[15];
  const float* ob4 = (const float*)d_in[16];
  const float* lw1 = (const float*)d_in[17];
  const float* lb1 = (const float*)d_in[18];
  const float* lw2 = (const float*)d_in[19];
  const float* lb2 = (const float*)d_in[20];
  float* out = (float*)d_out;

  char* ws = (char*)d_ws;
  half_t* h1 = (half_t*)ws;  ws += (size_t)4096 * 16 * 196 * 2;   // conv1 pooled f16
  half_t* h2 = (half_t*)ws;  ws += (size_t)4096 * 1568 * 2;       // flatten f16
  half_t* u1 = (half_t*)ws;  ws += (size_t)4096 * 128 * 2;        // enc hidden f16
  float*  z0 = (float*)ws;   ws += (size_t)4096 * 64 * 4;         // z0 f32
  float*  zT = (float*)ws;                                        // zT f32

  k_conv1<<<(4096 * 16 * 196) / 256, 256, 0, stream>>>(x, cw1, cb1, h1);
  k_conv2<<<(4096 * 32 * 49) / 256, 256, 0, stream>>>(h1, cw2, cb2, h2);
  k_lin1<<<32, 256, 0, stream>>>(h2, ew1, eb1, u1);
  k_lin2<<<32, 256, 0, stream>>>(u1, ew2, eb2, z0);

  // dynamic LDS: weights 96KB + z 16KB + k(6 slots) 96KB + staging 16KB = 224KB
  size_t lds = (size_t)49152 * 2 + 4 * 1024 * 4 + 4 * 6 * 1024 * 4 + 4 * 2048 * 2;
  k_ode<<<64, 128, lds, stream>>>(z0, zT, ow1, ob1, ow2, ob2, ow3, ob3, ow4, ob4);

  k_cls<<<16, 256, 0, stream>>>(zT, lw1, lb1, lw2, lb2, out);
}
